// RefinedNEXT_STAGE_G_29755533427481
// MI455X (gfx1250) — matrix operand order repaired and checked by reading, not compiled, not run on hardware
//
#include <hip/hip_runtime.h>
#include <hip/hip_bf16.h>
#include <math.h>

typedef __attribute__((ext_vector_type(16))) __bf16 v16bf;
typedef __attribute__((ext_vector_type(8)))  __bf16 v8bf;
typedef __attribute__((ext_vector_type(8)))  float  v8f;

#define TEMP_INV (1.0f/0.7f)
#define BN_RSQ   0.99999499987500096f  // 1/sqrt(1+1e-5)

// ---------------------------------------------------------------------------
// sourceT[b][o][s] = sum_c w_ctx[o][c] * word_embs[b][c][s]   (8,64,32)
// ---------------------------------------------------------------------------
__global__ __launch_bounds__(256) void k_sourceT(const float* __restrict__ wctx,
                                                 const float* __restrict__ we,
                                                 float* __restrict__ srcT) {
    int i = blockIdx.x * 256 + threadIdx.x;       // 16384 total
    int s = i & 31;
    int o = (i >> 5) & 63;
    int b = i >> 11;
    float acc = 0.f;
    for (int c = 0; c < 256; ++c)
        acc += wctx[o * 256 + c] * we[(b * 256 + c) * 32 + s];
    srcT[i] = acc;
}

// ---------------------------------------------------------------------------
// concat[:, 0:64] = h_code
// ---------------------------------------------------------------------------
__global__ __launch_bounds__(256) void k_copy_first_half(const float* __restrict__ h,
                                                         float* __restrict__ cat) {
    int i = blockIdx.x * 256 + threadIdx.x;       // 8*64*16384
    int q = i & 16383;
    int c = (i >> 14) & 63;
    int b = i >> 20;
    cat[(((size_t)b * 128 + c) << 14) + q] = h[i];
}

// ---------------------------------------------------------------------------
// Repack conv weights f32 -> bf16 in A-fragment staging order:
//   dst[((coTile*nCC + cc)*HALVES + h)*9*1024 + tap*1024 + co*32 + ci]
// so the conv kernel can bulk-copy one contiguous chunk per (coTile, cc).
// ---------------------------------------------------------------------------
__global__ __launch_bounds__(256) void k_repack(const float* __restrict__ Wt,
                                                __bf16* __restrict__ dst,
                                                int Cin, int CoutStore, int halves,
                                                int total) {
    int i = blockIdx.x * 256 + threadIdx.x;
    if (i >= total) return;
    int ci  = i & 31;
    int co  = (i >> 5) & 31;
    int tap = (i >> 10) % 9;
    int t   = (i >> 10) / 9;
    int h   = t % halves;
    int u   = t / halves;
    int nCC = Cin >> 5;
    int cc  = u % nCC;
    int coTile = u / nCC;
    int coG = coTile * 32 + co + h * CoutStore;
    int ciG = cc * 32 + ci;
    dst[i] = (__bf16)Wt[((size_t)coG * Cin + ciG) * 9 + tap];
}

// ---------------------------------------------------------------------------
// Attention: scores -> mask -> top-k(24) threshold -> softmax -> p
// ---------------------------------------------------------------------------
__global__ __launch_bounds__(256) void k_attn(const float* __restrict__ h_code,
                                              const float* __restrict__ srcT,
                                              const int*   __restrict__ mask,
                                              float* __restrict__ cat,
                                              float* __restrict__ att) {
    __shared__ float sT[64][32];
    __shared__ float msk[32];
    int b = blockIdx.x >> 6;
    int qBase = (blockIdx.x & 63) * 256;
    int tid = threadIdx.x;
    for (int i = tid; i < 2048; i += 256) sT[i >> 5][i & 31] = srcT[b * 2048 + i];
    if (tid < 32) msk[tid] = (mask[b * 32 + tid] != 0) ? 1.f : 0.f;
    __syncthreads();

    int q = qBase + tid;
    const float NEG = -__builtin_inff();

    float hc[64];
#pragma unroll 8
    for (int c = 0; c < 64; ++c)
        hc[c] = h_code[(((size_t)b * 64 + c) << 14) + q];

    float sc[32];
    for (int s = 0; s < 32; ++s) {
        float d = 0.f;
#pragma unroll 8
        for (int c = 0; c < 64; ++c) d += hc[c] * sT[c][s];
        sc[s] = (msk[s] != 0.f) ? NEG : d * TEMP_INV;
    }

    // threshold = 24th largest (selection with used-bitmask, duplicates OK)
    float thr = NEG;
    {
        unsigned used = 0u;
        for (int it = 0; it < 24; ++it) {
            int   bi = -1;
            float bv = 0.f;
            for (int s = 0; s < 32; ++s) {
                if (!(used & (1u << s)) && (bi < 0 || sc[s] > bv)) { bi = s; bv = sc[s]; }
            }
            used |= (1u << bi);
            thr = bv;
        }
    }
    for (int s = 0; s < 32; ++s) if (sc[s] < thr) sc[s] = NEG;

    float p[32];
    float mx = NEG;
    for (int s = 0; s < 32; ++s) mx = fmaxf(mx, sc[s]);
    if (mx == NEG) {
        for (int s = 0; s < 32; ++s) p[s] = 1.f / 32.f;
    } else {
        float sum = 0.f;
        for (int s = 0; s < 32; ++s) { float e = __expf(sc[s] - mx); p[s] = e; sum += e; }
        float inv = 1.f / sum;
        for (int s = 0; s < 32; ++s) p[s] *= inv;
    }

    for (int s = 0; s < 32; ++s)
        att[(((size_t)b * 32 + s) << 14) + q] = p[s];

    for (int o = 0; o < 64; ++o) {
        float acc = 0.f;
#pragma unroll 8
        for (int s = 0; s < 32; ++s) acc += sT[o][s] * p[s];
        cat[(((size_t)b * 128 + 64 + o) << 14) + q] = acc;
    }
}

// ---------------------------------------------------------------------------
// Fused conv3x3 (pad=1) + BN (+GLU | +residual) via bf16 WMMA implicit GEMM.
// Weights arrive pre-packed bf16 (k_repack) and are pulled into LDS with
// GLOBAL_LOAD_ASYNC_TO_LDS_B128 (ASYNCcnt), overlapping the VGPR-path input
// halo staging (transpose + pad + f32->bf16) that the async engine can't do.
// Block: 256 thr = 8 waves: 2 co-subtiles x 4 pixel-subtiles; tile = 32co x 64px.
// Grid: x = (W/64)*H, y = CoutStore/32, z = B.
// ---------------------------------------------------------------------------
template<int HALVES, bool UPSAMPLE, bool RESIDUAL, bool GLU>
__global__ __launch_bounds__(256)
void conv3x3_wmma(const float* __restrict__ X, const __bf16* __restrict__ Wp,
                  const float* __restrict__ gamma, const float* __restrict__ beta,
                  const float* __restrict__ R, float* __restrict__ O,
                  int Cin, int CoutStore, int H, int W, int inH, int inW) {
    __shared__ __align__(16) __bf16 Alds[HALVES][9][32][32]; // [half][tap][co][ci]
    __shared__ __align__(16) __bf16 Blds[3][66][32];         // [row][col][ci]

    const int tid  = threadIdx.x;
    const int lane = tid & 31;
    const int wave = tid >> 5;
    const int wrow = wave >> 2;                  // 0..1  (co subtile)
    const int wcol = wave & 3;                   // 0..3  (pixel subtile)
    const int coBlock = blockIdx.y * 32;
    const int b  = blockIdx.z;
    const int tilesX = W >> 6;
    const int xBase  = (blockIdx.x % tilesX) * 64;
    const int y      = blockIdx.x / tilesX;

    v8f acc[HALVES];
#pragma unroll
    for (int h = 0; h < HALVES; ++h) acc[h] = (v8f){0.f,0.f,0.f,0.f,0.f,0.f,0.f,0.f};

    const int m   = wrow * 16 + (lane & 15);
    const int akb = (lane < 16) ? 0 : 8;
    const int n   = wcol * 16 + (lane & 15);
    const int bkb = (lane < 16) ? 0 : 8;

    const unsigned aBase = (unsigned)(uintptr_t)(&Alds[0][0][0][0]); // LDS byte offset
    const int chunkElems = HALVES * 9 * 1024;                        // bf16 elems
    const int chunkBytes = chunkElems * 2;
    const int nChunks = Cin >> 5;

    for (int cc = 0; cc < nChunks; ++cc) {
        const int ciBase = cc << 5;

        // ---- async bulk-copy of pre-packed weight chunk into LDS (ASYNCcnt)
        {
            const char* wsrc = (const char*)(Wp +
                (size_t)(blockIdx.y * nChunks + cc) * chunkElems);
            for (int off = tid * 16; off < chunkBytes; off += 256 * 16) {
                unsigned    ldsaddr = aBase + (unsigned)off;
                const void* gaddr   = wsrc + off;
                asm volatile("global_load_async_to_lds_b128 %0, %1, off"
                             :: "v"(ldsaddr), "v"(gaddr) : "memory");
            }
        }
        // ---- stage input halo tile -> LDS (bf16), [row][col][ci] (VGPR path:
        //      needs transpose + zero pad + f32->bf16 conversion)
        for (int i = tid; i < 3 * 66 * 32; i += 256) {
            int col = i % 66;
            int r2  = i / 66;
            int ci  = r2 & 31;
            int row = r2 >> 5;
            int gy  = y + row - 1;
            int gx  = xBase + col - 1;
            float v = 0.f;
            if (gy >= 0 && gy < H && gx >= 0 && gx < W) {
                int sy = UPSAMPLE ? (gy >> 1) : gy;
                int sx = UPSAMPLE ? (gx >> 1) : gx;
                v = X[(((size_t)b * Cin + ciBase + ci) * inH + sy) * inW + sx];
            }
            Blds[row][col][ci] = (__bf16)v;
        }
        asm volatile("s_wait_asynccnt 0x0" ::: "memory");
        __syncthreads();

        // ---- 9 taps x HALVES WMMAs, K=32 bf16, f32 accumulate
#pragma unroll
        for (int ky = 0; ky < 3; ++ky) {
#pragma unroll
            for (int kx = 0; kx < 3; ++kx) {
                const int tap = ky * 3 + kx;
                v16bf bf;
                ((v8bf*)&bf)[0] = *(const v8bf*)&Blds[ky][n + kx][bkb];
                ((v8bf*)&bf)[1] = *(const v8bf*)&Blds[ky][n + kx][bkb + 16];
#pragma unroll
                for (int h = 0; h < HALVES; ++h) {
                    v16bf af;
                    ((v8bf*)&af)[0] = *(const v8bf*)&Alds[h][tap][m][akb];
                    ((v8bf*)&af)[1] = *(const v8bf*)&Alds[h][tap][m][akb + 16];
                    acc[h] = __builtin_amdgcn_wmma_f32_16x16x32_bf16(
                        false, af, false, bf, (short)0, acc[h], false, false);
                }
            }
        }
        __syncthreads();
    }

    // ---- epilogue: BN affine (scale = g/sqrt(1+eps)), GLU or residual
    const int coL = wrow * 16 + ((lane < 16) ? 0 : 8);
    const int px  = xBase + wcol * 16 + (lane & 15);
#pragma unroll
    for (int r = 0; r < 8; ++r) {
        const int co = coBlock + coL + r;
        float outv;
        if constexpr (GLU) {
            const int co1 = co + CoutStore;
            float a = acc[0][r] * (gamma[co]  * BN_RSQ) + beta[co];
            float g = acc[1][r] * (gamma[co1] * BN_RSQ) + beta[co1];
            outv = a * (1.f / (1.f + __expf(-g)));
        } else {
            outv = acc[0][r] * (gamma[co] * BN_RSQ) + beta[co];
            if constexpr (RESIDUAL)
                outv += R[(((size_t)b * CoutStore + co) * H + y) * W + px];
        }
        O[(((size_t)b * CoutStore + co) * H + y) * W + px] = outv;
    }
}

// ---------------------------------------------------------------------------
extern "C" void kernel_launch(void* const* d_in, const int* in_sizes, int n_in,
                              void* d_out, int out_size, void* d_ws, size_t ws_size,
                              hipStream_t stream) {
    const float* h_code    = (const float*)d_in[0];
    // d_in[1] = c_code (unused by reference output)
    const float* word_embs = (const float*)d_in[2];
    const int*   mask      = (const int*)  d_in[3];
    const float* w_ctx     = (const float*)d_in[4];
    const float* r0_w1 = (const float*)d_in[5];
    const float* r0_g1 = (const float*)d_in[6];
    const float* r0_b1 = (const float*)d_in[7];
    const float* r0_w2 = (const float*)d_in[8];
    const float* r0_g2 = (const float*)d_in[9];
    const float* r0_b2 = (const float*)d_in[10];
    const float* r1_w1 = (const float*)d_in[11];
    const float* r1_g1 = (const float*)d_in[12];
    const float* r1_b1 = (const float*)d_in[13];
    const float* r1_w2 = (const float*)d_in[14];
    const float* r1_g2 = (const float*)d_in[15];
    const float* r1_b2 = (const float*)d_in[16];
    const float* up_w  = (const float*)d_in[17];
    const float* up_g  = (const float*)d_in[18];
    const float* up_b  = (const float*)d_in[19];

    float* out = (float*)d_out;                       // (8,64,256,256)
    float* att = out + (size_t)8 * 64 * 256 * 256;    // (8,32,128,128)

    float* ws   = (float*)d_ws;
    float* srcT = ws;                                 // 8*64*32
    float* cat  = srcT + 16384;                       // (8,128,128,128)
    float* bufA = cat  + (size_t)16777216;
    float* bufB = bufA + (size_t)16777216;
    // packed bf16 weights after the f32 buffers
    __bf16* pw     = (__bf16*)(bufB + (size_t)16777216);
    __bf16* pw_r0w1 = pw;                             // 256*128*9 = 294912
    __bf16* pw_r0w2 = pw_r0w1 + 294912;               // 128*128*9 = 147456
    __bf16* pw_r1w1 = pw_r0w2 + 147456;
    __bf16* pw_r1w2 = pw_r1w1 + 294912;
    __bf16* pw_up   = pw_r1w2 + 147456;

    // 0. one-shot weight repack (f32 -> bf16, A-fragment order)
    k_repack<<<(294912 + 255) / 256, 256, 0, stream>>>(r0_w1, pw_r0w1, 128, 128, 2, 294912);
    k_repack<<<(147456 + 255) / 256, 256, 0, stream>>>(r0_w2, pw_r0w2, 128, 128, 1, 147456);
    k_repack<<<(294912 + 255) / 256, 256, 0, stream>>>(r1_w1, pw_r1w1, 128, 128, 2, 294912);
    k_repack<<<(147456 + 255) / 256, 256, 0, stream>>>(r1_w2, pw_r1w2, 128, 128, 1, 147456);
    k_repack<<<(147456 + 255) / 256, 256, 0, stream>>>(up_w,  pw_up,   128,  64, 2, 147456);

    // 1. sourceT = w_ctx @ word_embs
    k_sourceT<<<64, 256, 0, stream>>>(w_ctx, word_embs, srcT);
    // 2. concat first half = h_code
    k_copy_first_half<<<(8 * 64 * 16384) / 256, 256, 0, stream>>>(h_code, cat);
    // 3. attention -> att map + concat second half (wctx)
    k_attn<<<512, 256, 0, stream>>>(h_code, srcT, mask, cat, att);

    // 4. resblock 0
    conv3x3_wmma<2, false, false, true ><<<dim3(256, 4, 8), 256, 0, stream>>>(
        cat,  pw_r0w1, r0_g1, r0_b1, nullptr, bufA, 128, 128, 128, 128, 128, 128);
    conv3x3_wmma<1, false, true,  false><<<dim3(256, 4, 8), 256, 0, stream>>>(
        bufA, pw_r0w2, r0_g2, r0_b2, cat,     bufB, 128, 128, 128, 128, 128, 128);
    // 5. resblock 1
    conv3x3_wmma<2, false, false, true ><<<dim3(256, 4, 8), 256, 0, stream>>>(
        bufB, pw_r1w1, r1_g1, r1_b1, nullptr, bufA, 128, 128, 128, 128, 128, 128);
    conv3x3_wmma<1, false, true,  false><<<dim3(256, 4, 8), 256, 0, stream>>>(
        bufA, pw_r1w2, r1_g2, r1_b2, bufB,    cat,  128, 128, 128, 128, 128, 128);
    // 6. upsample2x fused into conv + BN + GLU -> final out
    conv3x3_wmma<2, true,  false, true ><<<dim3(1024, 2, 8), 256, 0, stream>>>(
        cat,  pw_up,   up_g,  up_b,  nullptr, out,  128, 64, 256, 256, 128, 128);
}
